// HSM3D_30305289240968
// MI455X (gfx1250) — matrix operand order repaired and checked by reading, not compiled, not run on hardware
//
#include <hip/hip_runtime.h>
#include <hip/hip_bf16.h>

#define N_PTS 500000
#define S_SP  10000
#define D_DIM 256
#define HID_  1024
#define K_SMP 10
#define TOK   (S_SP * (K_SMP + 1))   // 110000
#define TILE_M 64
#define MPAD  110080                  // TOK rounded up to a multiple of 128
#define CHUNK 64                      // hidden-dim chunk held in LDS

typedef __attribute__((ext_vector_type(16))) __bf16 bf16x16;
typedef __attribute__((ext_vector_type(8)))  float  f32x8;
typedef __attribute__((ext_vector_type(4)))  int    v4i;

#if defined(__AMDGCN__) && __has_builtin(__builtin_amdgcn_global_load_async_to_lds_b128) && \
    __has_builtin(__builtin_amdgcn_s_wait_asynccnt)
#define USE_ASYNC_LDS 1
#else
#define USE_ASYNC_LDS 0
#endif

#define AS1 __attribute__((address_space(1)))
#define AS3 __attribute__((address_space(3)))

// 16-byte global -> LDS copy; async (ASYNCcnt) when available on this toolchain
__device__ __forceinline__ void cp_b128(void* lds_dst, const void* gsrc) {
#if USE_ASYNC_LDS
    __builtin_amdgcn_global_load_async_to_lds_b128((AS1 v4i*)(void*)gsrc,
                                                   (AS3 v4i*)lds_dst, 0, 0);
#else
    *(uint4*)lds_dst = *(const uint4*)gsrc;
#endif
}
__device__ __forceinline__ void cp_wait() {
#if USE_ASYNC_LDS
    __builtin_amdgcn_s_wait_asynccnt(0);
#endif
}

__device__ __forceinline__ unsigned short f2bf(float f) {
    unsigned int u = __float_as_uint(f);
    unsigned int lsb = (u >> 16) & 1u;
    u += 0x7fffu + lsb;                 // round-to-nearest-even
    return (unsigned short)(u >> 16);
}
__device__ __forceinline__ unsigned int pack2(float a, float b) {
    return (unsigned int)f2bf(a) | ((unsigned int)f2bf(b) << 16);
}
__device__ __forceinline__ float gelu_f(float x) {
    float x3 = x * x * x;
    float t  = tanhf(0.7978845608028654f * (x + 0.044715f * x3));
    return 0.5f * x * (1.0f + t);
}

// ---- WMMA fragment loaders (ISA 7.12.2 layouts, wave32) -------------------
// A 16x32 bf16, row-major source with leading dim ld (elements).
__device__ __forceinline__ bf16x16 load_frag_a(const unsigned short* base, int ld, int lane) {
    int row = lane & 15;
    int k0  = (lane >> 4) * 8;
    union { uint4 q[2]; bf16x16 v; } f;
    const unsigned short* p = base + row * ld + k0;
    f.q[0] = *(const uint4*)(p);
    f.q[1] = *(const uint4*)(p + 16);
    return f.v;
}
// B 32x16 bf16 from N-major (pre-transposed) memory, leading dim ld.
__device__ __forceinline__ bf16x16 load_frag_b(const unsigned short* baseT, int ld, int lane) {
    int n  = lane & 15;
    int k0 = (lane >> 4) * 8;
    union { uint4 q[2]; bf16x16 v; } f;
    const unsigned short* p = baseT + n * ld + k0;
    f.q[0] = *(const uint4*)(p);
    f.q[1] = *(const uint4*)(p + 16);
    return f.v;
}

// ---------------------------------------------------------------------------
__global__ void zero_kernel(float* sums, int* cnt, int* cursor) {
    int i = blockIdx.x * blockDim.x + threadIdx.x;
    int stride = gridDim.x * blockDim.x;
    for (int j = i; j < S_SP * D_DIM; j += stride) sums[j] = 0.f;
    for (int j = i; j < S_SP; j += stride) { cnt[j] = 0; cursor[j] = 0; }
}

// one wave per point: segment-sum into sums[], count into cnt[]
__global__ void pool_kernel(const float* __restrict__ raw, const int* __restrict__ sp_idx,
                            float* __restrict__ sums, int* __restrict__ cnt) {
    int wave = (int)((blockIdx.x * blockDim.x + threadIdx.x) >> 5);
    int lane = threadIdx.x & 31;
    if (wave >= N_PTS) return;
    int sp = sp_idx[wave];
    const float* r = raw + (long)wave * D_DIM;
    float* s = sums + (long)sp * D_DIM;
#pragma unroll
    for (int j = 0; j < 8; j++) {
        int f = j * 32 + lane;
        atomicAdd(&s[f], r[f]);
    }
    if (lane == 0) atomicAdd(&cnt[sp], 1);
}

// single-block exclusive scan over counts -> starts
__global__ __launch_bounds__(1024) void scan_kernel(const int* __restrict__ cnt,
                                                    int* __restrict__ starts) {
    __shared__ int tmp[1024];
    __shared__ int carry;
    if (threadIdx.x == 0) carry = 0;
    __syncthreads();
    for (int base = 0; base < S_SP; base += 1024) {
        int i = base + (int)threadIdx.x;
        int v = (i < S_SP) ? cnt[i] : 0;
        tmp[threadIdx.x] = v;
        __syncthreads();
        for (int off = 1; off < 1024; off <<= 1) {
            int t = (threadIdx.x >= (unsigned)off) ? tmp[threadIdx.x - off] : 0;
            __syncthreads();
            tmp[threadIdx.x] += t;
            __syncthreads();
        }
        int incl = tmp[threadIdx.x];
        if (i < S_SP) starts[i] = carry + incl - v;
        __syncthreads();
        if (threadIdx.x == 1023) carry += tmp[1023];
        __syncthreads();
    }
}

__global__ void scatter_kernel(const int* __restrict__ sp_idx, const int* __restrict__ starts,
                               int* __restrict__ cursor, int* __restrict__ point_ids) {
    int i = blockIdx.x * blockDim.x + threadIdx.x;
    if (i >= N_PTS) return;
    int sp = sp_idx[i];
    int r = atomicAdd(&cursor[sp], 1);
    int pos = starts[sp] + r;
    if (pos < N_PTS) point_ids[pos] = i;
}

// w1 [D][HID] -> w1T bf16 [HID][D] ; w2 [HID][D] -> w2T bf16 [D][HID]
__global__ void wconv_kernel(const float* __restrict__ w1, const float* __restrict__ w2,
                             unsigned short* __restrict__ w1T, unsigned short* __restrict__ w2T) {
    int i = blockIdx.x * blockDim.x + threadIdx.x;
    if (i >= HID_ * D_DIM) return;
    int h = i / D_DIM, d = i - h * D_DIM;
    w1T[h * D_DIM + d] = f2bf(w1[d * HID_ + h]);
    w2T[d * HID_ + h]  = f2bf(w2[h * D_DIM + d]);
}

// one wave per token slot: build seq bf16 [MPAD][256]
__global__ void seq_kernel(const float* __restrict__ raw, const float* __restrict__ sums,
                           const int* __restrict__ cnt, const int* __restrict__ starts,
                           const int* __restrict__ point_ids, const int* __restrict__ rand_idx,
                           unsigned short* __restrict__ seq) {
    int tok  = (int)((blockIdx.x * blockDim.x + threadIdx.x) >> 5);
    int lane = threadIdx.x & 31;
    if (tok >= MPAD) return;
    unsigned short* dst = seq + (long)tok * D_DIM;
    int f0 = lane * 4, f1 = 128 + lane * 4;
    if (tok >= TOK) {
        *(uint2*)(dst + f0) = make_uint2(0u, 0u);
        *(uint2*)(dst + f1) = make_uint2(0u, 0u);
        return;
    }
    int s = tok / 11, j = tok - s * 11;
    if (j < K_SMP) {
        int c = cnt[s];
        int m = c > 0 ? c : 1;
        int ri = rand_idx[s * K_SMP + j] % m;
        int pos = starts[s] + ri;
        if (pos >= N_PTS) pos = N_PTS - 1;
        int src = point_ids[pos];
        float4 a = *(const float4*)(raw + (long)src * D_DIM + f0);
        float4 b = *(const float4*)(raw + (long)src * D_DIM + f1);
        *(uint2*)(dst + f0) = make_uint2(pack2(a.x, a.y), pack2(a.z, a.w));
        *(uint2*)(dst + f1) = make_uint2(pack2(b.x, b.y), pack2(b.z, b.w));
    } else {
        float inv = 1.f / fmaxf((float)cnt[s], 1.f);
        float4 a = *(const float4*)(sums + (long)s * D_DIM + f0);
        float4 b = *(const float4*)(sums + (long)s * D_DIM + f1);
        *(uint2*)(dst + f0) = make_uint2(pack2(a.x * inv, a.y * inv), pack2(a.z * inv, a.w * inv));
        *(uint2*)(dst + f1) = make_uint2(pack2(b.x * inv, b.y * inv), pack2(b.z * inv, b.w * inv));
    }
}

// Fused fc1 -> GELU -> fc2 with LDS-staged weights.
// 256 threads = 8 waves; 64-token tile per block; 104 KB LDS.
__global__ __launch_bounds__(256) void mlp_kernel(
    const unsigned short* __restrict__ seq,   // [MPAD][256] bf16
    const unsigned short* __restrict__ w1T,   // [HID][256]  bf16 (N-major for fc1 B)
    const unsigned short* __restrict__ w2T,   // [256][HID]  bf16 (N-major for fc2 B)
    const float* __restrict__ b1,             // [HID]
    const float* __restrict__ b2,             // [256]
    float* __restrict__ out)                  // [TOK][256]
{
    __shared__ __align__(16) unsigned short sA[TILE_M * D_DIM];   // 32 KB seq tile
    __shared__ __align__(16) unsigned short sH[TILE_M * CHUNK];   //  8 KB gelu'd hidden chunk
    __shared__ __align__(16) unsigned short w1c[CHUNK * D_DIM];   // 32 KB [h_local][256]
    __shared__ __align__(16) unsigned short w2c[D_DIM * CHUNK];   // 32 KB [out_col][64]

    int tid = threadIdx.x;
    int wave = tid >> 5, lane = tid & 31;
    long row0 = (long)blockIdx.x * TILE_M;

    int m4 = wave & 3;                  // M-subtile of this wave
    int nhalf = (wave >> 2) * 128;      // fc2 output-column half
    int nsub0 = (wave >> 2) * 2;        // fc1 N-subtile base within chunk (0..3)

    // seq tile: 2048 b128 transfers, 8 per thread (async to LDS)
    {
        const unsigned short* g = seq + row0 * D_DIM;
#pragma unroll
        for (int i = 0; i < 8; i++) {
            int e = (tid + i * 256) * 8;
            cp_b128(sA + e, g + e);
        }
    }

    // fc2 accumulators, initialized with b2
    f32x8 acc2[8];
#pragma unroll
    for (int ns = 0; ns < 8; ns++) {
        float bv = b2[nhalf + ns * 16 + (lane & 15)];
#pragma unroll
        for (int r = 0; r < 8; r++) acc2[ns][r] = bv;
    }

    for (int hk = 0; hk < HID_ / CHUNK; hk++) {   // 16 chunks
        // ---- stage weight chunks into LDS (async) ----
        {
            const unsigned short* g = w1T + (long)hk * CHUNK * D_DIM;  // contiguous 32 KB
#pragma unroll
            for (int i = 0; i < 8; i++) {
                int e = (tid + i * 256) * 8;
                cp_b128(w1c + e, g + e);
            }
#pragma unroll
            for (int i = 0; i < 8; i++) {          // strided rows of w2T
                int t = tid + i * 256;
                int rowc = t >> 3;                 // out col 0..255
                int seg  = t & 7;                  // 16B segment within 128B row
                cp_b128(w2c + rowc * CHUNK + seg * 8,
                        w2T + (long)rowc * HID_ + hk * CHUNK + seg * 8);
            }
        }
        cp_wait();
        __syncthreads();

        // ---- fc1: h_sub[64 x 64] = seq_tile @ w1c, +b1, GELU -> sH bf16 ----
        f32x8 acc1[2] = { {}, {} };
#pragma unroll
        for (int kk = 0; kk < 8; kk++) {           // K over D=256
            bf16x16 a = load_frag_a(sA + (m4 * 16) * D_DIM + kk * 32, D_DIM, lane);
#pragma unroll
            for (int nn = 0; nn < 2; nn++) {
                bf16x16 b = load_frag_b(w1c + (nsub0 + nn) * 16 * D_DIM + kk * 32, D_DIM, lane);
                acc1[nn] = __builtin_amdgcn_wmma_f32_16x16x32_bf16(false, a, false, b,
                                                                   (short)0, acc1[nn], false, false);
            }
        }
#pragma unroll
        for (int nn = 0; nn < 2; nn++) {
            int ncl = (nsub0 + nn) * 16 + (lane & 15);
            float b1v = b1[hk * CHUNK + ncl];
#pragma unroll
            for (int r = 0; r < 8; r++) {
                int mrow = m4 * 16 + r + ((lane >> 4) * 8);
                sH[mrow * CHUNK + ncl] = f2bf(gelu_f(acc1[nn][r] + b1v));
            }
        }
        __syncthreads();

        // ---- fc2: acc2 += h_chunk @ w2c ----
#pragma unroll
        for (int kk = 0; kk < 2; kk++) {           // K over CHUNK=64
            bf16x16 a = load_frag_a(sH + (m4 * 16) * CHUNK + kk * 32, CHUNK, lane);
#pragma unroll
            for (int ns = 0; ns < 8; ns++) {
                bf16x16 b = load_frag_b(w2c + (nhalf + ns * 16) * CHUNK + kk * 32, CHUNK, lane);
                acc2[ns] = __builtin_amdgcn_wmma_f32_16x16x32_bf16(false, a, false, b,
                                                                   (short)0, acc2[ns], false, false);
            }
        }
        __syncthreads();   // protect sH / w1c / w2c before next chunk
    }

    // store D: element (M = r + 8*(lane>>4), N = lane&15) per 16x16 subtile
#pragma unroll
    for (int ns = 0; ns < 8; ns++) {
        int ncol = nhalf + ns * 16 + (lane & 15);
#pragma unroll
        for (int r = 0; r < 8; r++) {
            long trow = row0 + m4 * 16 + r + ((lane >> 4) * 8);
            if (trow < (long)TOK) out[trow * D_DIM + ncol] = acc2[ns][r];
        }
    }
}

// ---------------------------------------------------------------------------
extern "C" void kernel_launch(void* const* d_in, const int* in_sizes, int n_in,
                              void* d_out, int out_size, void* d_ws, size_t ws_size,
                              hipStream_t stream) {
    const float* raw      = (const float*)d_in[0];
    const float* w1       = (const float*)d_in[1];
    const float* b1       = (const float*)d_in[2];
    const float* w2       = (const float*)d_in[3];
    const float* b2       = (const float*)d_in[4];
    const int*   sp_idx   = (const int*)d_in[5];
    const int*   rand_idx = (const int*)d_in[6];
    float* out = (float*)d_out;

    char* p = (char*)d_ws;
    auto alloc = [&](size_t bytes) -> void* {
        void* r = (void*)p;
        p += (bytes + 255) & ~(size_t)255;
        return r;
    };
    float*          sums      = (float*)alloc((size_t)S_SP * D_DIM * sizeof(float));
    int*            cnt       = (int*)alloc((size_t)S_SP * sizeof(int));
    int*            starts    = (int*)alloc((size_t)S_SP * sizeof(int));
    int*            cursor    = (int*)alloc((size_t)S_SP * sizeof(int));
    int*            point_ids = (int*)alloc((size_t)N_PTS * sizeof(int));
    unsigned short* seqb      = (unsigned short*)alloc((size_t)MPAD * D_DIM * sizeof(unsigned short));
    unsigned short* w1T       = (unsigned short*)alloc((size_t)HID_ * D_DIM * sizeof(unsigned short));
    unsigned short* w2T       = (unsigned short*)alloc((size_t)D_DIM * HID_ * sizeof(unsigned short));

    zero_kernel<<<512, 256, 0, stream>>>(sums, cnt, cursor);
    pool_kernel<<<N_PTS / 8, 256, 0, stream>>>(raw, sp_idx, sums, cnt);
    scan_kernel<<<1, 1024, 0, stream>>>(cnt, starts);
    scatter_kernel<<<(N_PTS + 255) / 256, 256, 0, stream>>>(sp_idx, starts, cursor, point_ids);
    wconv_kernel<<<(HID_ * D_DIM + 255) / 256, 256, 0, stream>>>(w1, w2, w1T, w2T);
    seq_kernel<<<MPAD / 8, 256, 0, stream>>>(raw, sums, cnt, starts, point_ids, rand_idx, seqb);
    mlp_kernel<<<MPAD / TILE_M, 256, 0, stream>>>(seqb, w1T, w2T, b1, b2, out);
}
